// SelectiveSSM_3444563771624
// MI455X (gfx1250) — matrix operand order repaired and checked by reading, not compiled, not run on hardware
//
#include <hip/hip_runtime.h>

// ---------------- problem constants (match reference) ----------------
#define D_INNER 2048
#define D_STATE 16
#define DT_RANK 128
#define KPROJ   160            // DT_RANK + 2*D_STATE
#define BATCH   2
#define SEQLEN  2048
#define M_TOT   (BATCH*SEQLEN) // 4096 rows for both GEMMs
#define NCH     8              // scan chunks per sequence
#define CHUNK   (SEQLEN/NCH)   // 256 steps per chunk

// ---------------- WMMA types ----------------
typedef __attribute__((ext_vector_type(16))) __bf16 v16bf;
typedef __attribute__((ext_vector_type(8)))  float  v8f;
typedef __attribute__((ext_vector_type(4)))  unsigned int v4u;

struct U32x8 { v4u lo; v4u hi; };   // 32 bytes == v16bf

__device__ __forceinline__ unsigned short f2bf(float f) {
  unsigned int u = __float_as_uint(f);
  u += 0x7FFFu + ((u >> 16) & 1u);     // round-to-nearest-even
  return (unsigned short)(u >> 16);
}

// A fragment: 16x32 bf16 tile, row-major source (lda = K stride).
// Lane l: row m = l%16; K-groups [h*8, h*8+8) and [16+h*8, 16+h*8+8), h = l/16.
__device__ __forceinline__ v16bf load_fragA(const unsigned short* A, int lda,
                                            int m0, int k0, int lane) {
  int h = lane >> 4, r = lane & 15;
  const unsigned short* p = A + (size_t)(m0 + r) * lda + k0 + h * 8;
  U32x8 u;
  u.lo = *(const v4u*)(p);        // 8 bf16 = 16B
  u.hi = *(const v4u*)(p + 16);   // +16 elements
  return __builtin_bit_cast(v16bf, u);
}

// B fragment: 32x16 bf16 tile loaded from B^T (N x K row-major, ldb = K stride).
// Lane l: col n = l%16; K-groups [h*8, h*8+8) and [16+h*8, 16+h*8+8), h = l/16.
__device__ __forceinline__ v16bf load_fragB(const unsigned short* BT, int ldb,
                                            int n0, int k0, int lane) {
  int h = lane >> 4, n = lane & 15;
  const unsigned short* p = BT + (size_t)(n0 + n) * ldb + k0 + h * 8;
  U32x8 u;
  u.lo = *(const v4u*)(p);
  u.hi = *(const v4u*)(p + 16);
  return __builtin_bit_cast(v16bf, u);
}

// D store: lane l, vgpr v -> (m = h*8+v, n = l%16)
__device__ __forceinline__ void store_tile(float* C, int ldc, int m0, int n0,
                                           int lane, v8f acc) {
  int h = lane >> 4, n = lane & 15;
#pragma unroll
  for (int v = 0; v < 8; ++v)
    C[(size_t)(m0 + h * 8 + v) * ldc + n0 + n] = acc[v];
}

// ---------------- conversion kernels ----------------
__global__ void k_cvt_bf16(const float* __restrict__ in,
                           unsigned short* __restrict__ out, int n) {
  int i = blockIdx.x * blockDim.x + threadIdx.x;
  if (i < n) out[i] = f2bf(in[i]);
}

// first DT_RANK columns of x_dbl (ld=KPROJ) -> bf16 (ld=DT_RANK)
__global__ void k_cvt_dun(const float* __restrict__ xdbl,
                          unsigned short* __restrict__ dun) {
  int i = blockIdx.x * blockDim.x + threadIdx.x;      // 0 .. M_TOT*DT_RANK-1
  if (i < M_TOT * DT_RANK) {
    int m = i / DT_RANK, r = i % DT_RANK;
    dun[i] = f2bf(xdbl[(size_t)m * KPROJ + r]);
  }
}

// ---------------- GEMM1: x_dbl = x @ x_proj_w^T  (M=4096,K=2048,N=160) ------
// 10 waves/block, each wave owns one 16-wide N tile; block owns one 16-row M tile.
__global__ void k_gemm1(const unsigned short* __restrict__ Ab,
                        const unsigned short* __restrict__ W1b,
                        float* __restrict__ xdbl) {
  int lane = threadIdx.x & 31;
  int wave = threadIdx.x >> 5;          // 0..9
  int m0 = blockIdx.x * 16;
  int n0 = wave * 16;
  v8f acc = {};
  for (int k0 = 0; k0 < D_INNER; k0 += 32) {
    v16bf a = load_fragA(Ab, D_INNER, m0, k0, lane);
    v16bf b = load_fragB(W1b, D_INNER, n0, k0, lane);
    acc = __builtin_amdgcn_wmma_f32_16x16x32_bf16(false, a, false, b,
                                                  (short)0, acc, false, false);
  }
  store_tile(xdbl, KPROJ, m0, n0, lane, acc);
}

// ---- GEMM2 + bias + softplus: delta = softplus(delta_un @ dt_proj_w^T + b) --
// (M=4096, K=128, N=2048); 8 waves/block, one 16x16 tile per wave.
__global__ void k_gemm2(const unsigned short* __restrict__ dunb,
                        const unsigned short* __restrict__ W2b,
                        const float* __restrict__ bias,
                        float* __restrict__ delta) {
  int lane = threadIdx.x & 31;
  int wave = threadIdx.x >> 5;          // 0..7
  int m0 = blockIdx.x * 16;
  int n0 = (blockIdx.y * 8 + wave) * 16;
  v8f acc = {};
#pragma unroll
  for (int k0 = 0; k0 < DT_RANK; k0 += 32) {
    v16bf a = load_fragA(dunb, DT_RANK, m0, k0, lane);
    v16bf b = load_fragB(W2b, DT_RANK, n0, k0, lane);
    acc = __builtin_amdgcn_wmma_f32_16x16x32_bf16(false, a, false, b,
                                                  (short)0, acc, false, false);
  }
  int h = lane >> 4, n = lane & 15;
  float bn = bias[n0 + n];
#pragma unroll
  for (int v = 0; v < 8; ++v) {
    float z = acc[v] + bn;
    float sp = (z > 20.f) ? z : log1pf(__expf(z));
    delta[(size_t)(m0 + h * 8 + v) * D_INNER + n0 + n] = sp;
  }
}

// ---------------- scan phase 1: per-chunk summaries (P = prod a, S) ---------
__global__ void k_scan1(const float* __restrict__ x,
                        const float* __restrict__ xdbl,
                        const float* __restrict__ delta,
                        const float* __restrict__ A_log,
                        float* __restrict__ P, float* __restrict__ S) {
  __shared__ float sB[CHUNK * D_STATE];           // 16 KB
  int b = blockIdx.z, c = blockIdx.y;
  int d = blockIdx.x * 256 + threadIdx.x;
  for (int i = threadIdx.x; i < CHUNK * D_STATE; i += 256) {
    int l = i >> 4, n = i & 15;
    sB[i] = xdbl[(size_t)(b * SEQLEN + c * CHUNK + l) * KPROJ + DT_RANK + n];
  }
  __syncthreads();

  float An[D_STATE], Pr[D_STATE], Sr[D_STATE];
#pragma unroll
  for (int n = 0; n < D_STATE; ++n) {
    An[n] = -__expf(A_log[(size_t)d * D_STATE + n]);
    Pr[n] = 1.f; Sr[n] = 0.f;
  }
  for (int l = 0; l < CHUNK; ++l) {
    size_t row = (size_t)(b * SEQLEN + c * CHUNK + l);
    float dt = delta[row * D_INNER + d];
    float xv = x[row * D_INNER + d];
    if (l + 8 < CHUNK) {
      __builtin_prefetch(&delta[(row + 8) * D_INNER + d], 0, 0);
      __builtin_prefetch(&x[(row + 8) * D_INNER + d], 0, 0);
    }
    float dx = dt * xv;
#pragma unroll
    for (int n = 0; n < D_STATE; ++n) {
      float a = __expf(dt * An[n]);
      float u = dx * sB[l * D_STATE + n];
      Pr[n] *= a;
      Sr[n] = a * Sr[n] + u;
    }
  }
  size_t base = ((size_t)(b * NCH + c) * D_INNER + d) * D_STATE;
#pragma unroll
  for (int n = 0; n < D_STATE; ++n) { P[base + n] = Pr[n]; S[base + n] = Sr[n]; }
}

// ---------------- scan phase 2: prefix-combine chunk summaries --------------
__global__ void k_scan2(const float* __restrict__ P, const float* __restrict__ S,
                        float* __restrict__ init) {
  int idx = blockIdx.x * 256 + threadIdx.x;       // 0 .. B*D-1
  int b = idx >> 11, d = idx & (D_INNER - 1);
  float s[D_STATE];
#pragma unroll
  for (int n = 0; n < D_STATE; ++n) s[n] = 0.f;
  for (int c = 0; c < NCH; ++c) {
    size_t base = ((size_t)(b * NCH + c) * D_INNER + d) * D_STATE;
#pragma unroll
    for (int n = 0; n < D_STATE; ++n) init[base + n] = s[n];
#pragma unroll
    for (int n = 0; n < D_STATE; ++n) s[n] = P[base + n] * s[n] + S[base + n];
  }
}

// ---------------- scan phase 3: replay chunks, emit y -----------------------
__global__ void k_scan3(const float* __restrict__ x,
                        const float* __restrict__ xdbl,
                        const float* __restrict__ delta,
                        const float* __restrict__ A_log,
                        const float* __restrict__ Dp,
                        const float* __restrict__ init,
                        float* __restrict__ out) {
  __shared__ float sB[CHUNK * D_STATE];           // 16 KB
  __shared__ float sC[CHUNK * D_STATE];           // 16 KB
  int b = blockIdx.z, c = blockIdx.y;
  int d = blockIdx.x * 256 + threadIdx.x;
  for (int i = threadIdx.x; i < CHUNK * D_STATE; i += 256) {
    int l = i >> 4, n = i & 15;
    size_t row = (size_t)(b * SEQLEN + c * CHUNK + l);
    sB[i] = xdbl[row * KPROJ + DT_RANK + n];
    sC[i] = xdbl[row * KPROJ + DT_RANK + D_STATE + n];
  }
  __syncthreads();

  float An[D_STATE], st[D_STATE];
  size_t base = ((size_t)(b * NCH + c) * D_INNER + d) * D_STATE;
#pragma unroll
  for (int n = 0; n < D_STATE; ++n) {
    An[n] = -__expf(A_log[(size_t)d * D_STATE + n]);
    st[n] = init[base + n];
  }
  float dpar = Dp[d];

  for (int l = 0; l < CHUNK; ++l) {
    size_t row = (size_t)(b * SEQLEN + c * CHUNK + l);
    float dt = delta[row * D_INNER + d];
    float xv = x[row * D_INNER + d];
    if (l + 8 < CHUNK) {
      __builtin_prefetch(&delta[(row + 8) * D_INNER + d], 0, 0);
      __builtin_prefetch(&x[(row + 8) * D_INNER + d], 0, 0);
    }
    float dx = dt * xv;
    float y = 0.f;
#pragma unroll
    for (int n = 0; n < D_STATE; ++n) {
      float a = __expf(dt * An[n]);
      float u = dx * sB[l * D_STATE + n];
      st[n] = a * st[n] + u;
      y += st[n] * sC[l * D_STATE + n];
    }
    out[row * D_INNER + d] = y + xv * dpar;
  }
}

// ---------------- host launcher ----------------
extern "C" void kernel_launch(void* const* d_in, const int* in_sizes, int n_in,
                              void* d_out, int out_size, void* d_ws, size_t ws_size,
                              hipStream_t stream) {
  const float* x    = (const float*)d_in[0];   // (B,L,D)
  const float* w1   = (const float*)d_in[1];   // (KPROJ, D)
  const float* w2   = (const float*)d_in[2];   // (D, DT_RANK)
  const float* dtb  = (const float*)d_in[3];   // (D,)
  const float* alog = (const float*)d_in[4];   // (D, N)
  const float* dpar = (const float*)d_in[5];   // (D,)
  float* out = (float*)d_out;

  char* ws = (char*)d_ws;
  size_t off = 0;
  auto take = [&](size_t bytes) {
    size_t o = off;
    off = (off + bytes + 255) & ~(size_t)255;
    return o;
  };
  unsigned short* xb   = (unsigned short*)(ws + take((size_t)M_TOT * D_INNER * 2));
  unsigned short* w1b  = (unsigned short*)(ws + take((size_t)KPROJ * D_INNER * 2));
  unsigned short* w2b  = (unsigned short*)(ws + take((size_t)D_INNER * DT_RANK * 2));
  unsigned short* dunb = (unsigned short*)(ws + take((size_t)M_TOT * DT_RANK * 2));
  float* xdbl  = (float*)(ws + take((size_t)M_TOT * KPROJ * 4));
  float* delta = (float*)(ws + take((size_t)M_TOT * D_INNER * 4));
  float* P     = (float*)(ws + take((size_t)BATCH * NCH * D_INNER * D_STATE * 4));
  float* S     = (float*)(ws + take((size_t)BATCH * NCH * D_INNER * D_STATE * 4));
  float* initS = (float*)(ws + take((size_t)BATCH * NCH * D_INNER * D_STATE * 4));
  (void)ws_size; (void)in_sizes; (void)n_in; (void)out_size;

  // 1) bf16 conversions
  {
    int n = M_TOT * D_INNER;
    k_cvt_bf16<<<(n + 255) / 256, 256, 0, stream>>>(x, xb, n);
    n = KPROJ * D_INNER;
    k_cvt_bf16<<<(n + 255) / 256, 256, 0, stream>>>(w1, w1b, n);
    n = D_INNER * DT_RANK;
    k_cvt_bf16<<<(n + 255) / 256, 256, 0, stream>>>(w2, w2b, n);
  }
  // 2) GEMM1: x_dbl = x @ w1^T
  k_gemm1<<<M_TOT / 16, 320, 0, stream>>>(xb, w1b, xdbl);
  // 3) delta_un -> bf16
  {
    int n = M_TOT * DT_RANK;
    k_cvt_dun<<<(n + 255) / 256, 256, 0, stream>>>(xdbl, dunb);
  }
  // 4) GEMM2 + bias + softplus -> delta
  k_gemm2<<<dim3(M_TOT / 16, D_INNER / 128), 256, 0, stream>>>(dunb, w2b, dtb, delta);
  // 5) chunked scan
  dim3 sg(D_INNER / 256, NCH, BATCH);
  k_scan1<<<sg, 256, 0, stream>>>(x, xdbl, delta, alog, P, S);
  k_scan2<<<(BATCH * D_INNER) / 256, 256, 0, stream>>>(P, S, initS);
  k_scan3<<<sg, 256, 0, stream>>>(x, xdbl, delta, alog, dpar, initS, out);
}
